// bGCN_27410481283416
// MI455X (gfx1250) — matrix operand order repaired and checked by reading, not compiled, not run on hardware
//
#include <hip/hip_runtime.h>

// ---------------------------------------------------------------------------
// bGCN forward for MI455X (gfx1250, wave32, WMMA 16x16x32 f16 -> f32 acc)
//
//  Kernel 1: Zc = V @ Wvc  (-> d_out),  Vn = V @ Wvn (-> d_ws)
//    - block = 256 threads = 8 waves, covers 128 rows of V
//    - each wave owns one unique 16-row tile -> A read exactly once from HBM
//    - both weight matrices staged once per block into LDS (f16, column-major)
//    - fast path (full 16-row tile): fully unrolled, double-buffered B
//      fragments (bA/bB) so ds_load_b128 of tile t+1 overlap the WMMAs of
//      tile t; unconditional clause stores with immediate offsets
//  Kernel 2: out = relu(Zc + (sum_k e_k * Vn[idx_k]) / cnt + bv)
//    - Vn (25.6 MB) is L2-resident (192 MB L2) -> gather never touches HBM
// ---------------------------------------------------------------------------

typedef _Float16 v16h __attribute__((ext_vector_type(16)));
typedef _Float16 v8h  __attribute__((ext_vector_type(8)));
typedef float    v8f  __attribute__((ext_vector_type(8)));

#define FEATS       128
#define LDS_STRIDE  136                  // padded K-stride (halves): breaks bank-conflict period
#define MAT_OFF     (FEATS * LDS_STRIDE) // halves per staged weight matrix

extern __shared__ _Float16 smemW[];      // 2 * 128 * 136 * 2B = 69632 B dynamic LDS

__global__ __launch_bounds__(256)
void gemm_dual_kernel(const float* __restrict__ vertices,
                      const float* __restrict__ Wvc,
                      const float* __restrict__ Wvn,
                      float* __restrict__ Zc,   // d_out
                      float* __restrict__ Vn,   // d_ws
                      int N)
{
    const int tid = threadIdx.x;

    // ---- Stage both weight matrices into LDS as f16, column-major (K contiguous)
    {
        const float4* srcs[2] = { (const float4*)Wvc, (const float4*)Wvn };
        #pragma unroll
        for (int m = 0; m < 2; ++m) {
            _Float16* dst = smemW + m * MAT_OFF;
            for (int i = tid; i < (FEATS * FEATS) / 4; i += 256) {
                float4 x = srcs[m][i];
                int k = i >> 5;            // row (K index) of W
                int c = (i & 31) << 2;     // starting column
                dst[(c + 0) * LDS_STRIDE + k] = (_Float16)x.x;
                dst[(c + 1) * LDS_STRIDE + k] = (_Float16)x.y;
                dst[(c + 2) * LDS_STRIDE + k] = (_Float16)x.z;
                dst[(c + 3) * LDS_STRIDE + k] = (_Float16)x.w;
            }
        }
    }
    __syncthreads();

    const int lane = tid & 31;
    const int wave = tid >> 5;
    const int half = lane >> 4;   // selects K sub-range per ISA A/B layouts
    const int m16  = lane & 15;   // A: row M; B/C/D: column N

    // Each wave owns one unique 16-row tile of the 128-row block panel.
    const int rowTile = blockIdx.x * 128 + wave * 16;
    int rowA = rowTile + m16;
    if (rowA > N - 1) rowA = N - 1;            // clamp (EXEC must stay all-1 for WMMA)
    const float* Ar = vertices + (size_t)rowA * FEATS;

    // ---- A fragments for all 4 K-blocks (K = 128 = 4 * 32)
    // A 16x32 layout (matches B): elem j -> K = 16*half + j
    v16h a[4];
    #pragma unroll
    for (int kb = 0; kb < 4; ++kb) {
        const int k0 = kb * 32 + half * 16;
        const int k1 = kb * 32 + half * 16 + 8;
        float4 x0 = *(const float4*)(Ar + k0);
        float4 x1 = *(const float4*)(Ar + k0 + 4);
        float4 y0 = *(const float4*)(Ar + k1);
        float4 y1 = *(const float4*)(Ar + k1 + 4);
        a[kb][0]  = (_Float16)x0.x; a[kb][1]  = (_Float16)x0.y;
        a[kb][2]  = (_Float16)x0.z; a[kb][3]  = (_Float16)x0.w;
        a[kb][4]  = (_Float16)x1.x; a[kb][5]  = (_Float16)x1.y;
        a[kb][6]  = (_Float16)x1.z; a[kb][7]  = (_Float16)x1.w;
        a[kb][8]  = (_Float16)y0.x; a[kb][9]  = (_Float16)y0.y;
        a[kb][10] = (_Float16)y0.z; a[kb][11] = (_Float16)y0.w;
        a[kb][12] = (_Float16)y1.x; a[kb][13] = (_Float16)y1.y;
        a[kb][14] = (_Float16)y1.z; a[kb][15] = (_Float16)y1.w;
    }

    // Per-wave bases.
    const _Float16* wLane = smemW + m16 * LDS_STRIDE + half * 16;
    const size_t    oBase = (size_t)(rowTile + half * 8) * FEATS + m16;

    // B 32x16 layout: elem j -> K = 32*kb + 16*half + j, column = m16.
    // Column-major LDS => 16 contiguous halves per fragment (2x ds_load_b128).
    auto loadB = [&](v16h* b, const _Float16* wMat, int ct) {
        const _Float16* bp = wMat + ct * (16 * LDS_STRIDE);
        #pragma unroll
        for (int kb = 0; kb < 4; ++kb) {
            const _Float16* p = bp + kb * 32;
            v8h lo = *(const v8h*)(p);
            v8h hi = *(const v8h*)(p + 8);
            #pragma unroll
            for (int j = 0; j < 8; ++j) { b[kb][j] = lo[j]; b[kb][8 + j] = hi[j]; }
        }
    };

    auto tileCompute = [&](const v16h* b) -> v8f {
        v8f acc = {0.f, 0.f, 0.f, 0.f, 0.f, 0.f, 0.f, 0.f};
        #pragma unroll
        for (int kb = 0; kb < 4; ++kb) {
            acc = __builtin_amdgcn_wmma_f32_16x16x32_f16(
                      /*neg_a=*/false, a[kb], /*neg_b=*/false, b[kb],
                      /*c_mod=*/(short)0, acc, /*reuse_a=*/false, /*reuse_b=*/false);
        }
        return acc;
    };

    if (rowTile + 16 <= N) {
        // ================= fast path: branch-free, double-buffered =========
        #pragma unroll
        for (int mat = 0; mat < 2; ++mat) {
            const _Float16* wMat = wLane + mat * MAT_OFF;
            float* outBase = ((mat == 0) ? Zc : Vn) + oBase;

            v16h bA[4], bB[4];
            loadB(bA, wMat, 0);
            #pragma unroll
            for (int ct = 0; ct < 8; ct += 2) {
                // issue next tile's 8 ds_load_b128 before consuming bA
                loadB(bB, wMat, ct + 1);
                v8f acc0 = tileCompute(bA);
                #pragma unroll
                for (int i = 0; i < 8; ++i)
                    outBase[(size_t)i * FEATS + ct * 16] = acc0[i];

                if (ct + 2 < 8) loadB(bA, wMat, ct + 2);
                v8f acc1 = tileCompute(bB);
                #pragma unroll
                for (int i = 0; i < 8; ++i)
                    outBase[(size_t)i * FEATS + (ct + 1) * 16] = acc1[i];
            }
        }
    } else {
        // ================= tail path: only the last partial block ==========
        #pragma unroll
        for (int mat = 0; mat < 2; ++mat) {
            const _Float16* wMat = wLane + mat * MAT_OFF;
            float* outBase = ((mat == 0) ? Zc : Vn) + oBase;

            for (int ct = 0; ct < 8; ++ct) {
                v16h b[4];
                loadB(b, wMat, ct);
                v8f acc = tileCompute(b);
                #pragma unroll
                for (int i = 0; i < 8; ++i)
                    if (rowTile + half * 8 + i < N)
                        outBase[(size_t)i * FEATS + ct * 16] = acc[i];
            }
        }
    }
}

// ---------------------------------------------------------------------------
// Gather-aggregate epilogue: 2 vertices per block, 1 thread per feature.
// ---------------------------------------------------------------------------
__global__ __launch_bounds__(256)
void gather_agg_kernel(const int*   __restrict__ nh_idx,
                       const int*   __restrict__ int_idx,
                       const float* __restrict__ nh_e,
                       const float* __restrict__ int_e,
                       const float* __restrict__ Vn,
                       const float* __restrict__ bv,
                       float* __restrict__ out,   // in: Zc, out: z
                       int N)
{
    __shared__ int   sIdx[2][40];
    __shared__ float sE[2][40];

    const int sub = threadIdx.x >> 7;   // vertex slot within block (0..1)
    const int f   = threadIdx.x & 127;  // feature
    const int v   = blockIdx.x * 2 + sub;

    if (v < N && f < 40) {
        int idx; float e;
        if (f < 20) { idx = nh_idx[v * 20 + f];        e = nh_e[v * 20 + f]; }
        else        { idx = int_idx[v * 20 + f - 20];  e = int_e[v * 20 + f - 20]; }
        sIdx[sub][f] = idx;
        sE[sub][f]   = e;
    }
    __syncthreads();
    if (v >= N) return;

    float acc = 0.f;
    int   cnt = 0;
    #pragma unroll 4
    for (int k = 0; k < 40; ++k) {
        const int idx = sIdx[sub][k];
        if (idx >= 0) {
            ++cnt;
            acc = fmaf(sE[sub][k], Vn[(size_t)idx * FEATS + f], acc);
        }
    }

    const size_t o = (size_t)v * FEATS + f;
    float z = out[o] + acc / (float)cnt + bv[f];
    out[o] = z > 0.f ? z : 0.f;
}

// ---------------------------------------------------------------------------
extern "C" void kernel_launch(void* const* d_in, const int* in_sizes, int n_in,
                              void* d_out, int out_size, void* d_ws, size_t ws_size,
                              hipStream_t stream)
{
    const float* vertices = (const float*)d_in[0];
    const int*   nh_idx   = (const int*)  d_in[1];
    const int*   int_idx  = (const int*)  d_in[2];
    const float* nh_e     = (const float*)d_in[3];
    const float* int_e    = (const float*)d_in[4];
    const float* Wvc      = (const float*)d_in[5];
    const float* Wvn      = (const float*)d_in[6];
    const float* bv       = (const float*)d_in[7];

    float* out = (float*)d_out;          // Zc, then final z
    float* Vn  = (float*)d_ws;           // N*128 floats of scratch

    const int N = in_sizes[0] / FEATS;

    const int    gemmBlocks = (N + 127) / 128;                // 8 waves x 16 rows
    const size_t ldsBytes   = 2u * FEATS * LDS_STRIDE * 2u;   // 69632 B

    gemm_dual_kernel<<<gemmBlocks, 256, ldsBytes, stream>>>(
        vertices, Wvc, Wvn, out, Vn, N);

    const int aggBlocks = (N + 1) / 2;
    gather_agg_kernel<<<aggBlocks, 256, 0, stream>>>(
        nh_idx, int_idx, nh_e, int_e, Vn, bv, out, N);
}
